// MechanismGrabber_28887950033396
// MI455X (gfx1250) — matrix operand order repaired and checked by reading, not compiled, not run on hardware
//
#include <hip/hip_runtime.h>
#include <hip/hip_bf16.h>
#include <math.h>

// MI455X / gfx1250, wave32. bf16 WMMA 16x16x32, f32 accumulate.

typedef __attribute__((ext_vector_type(16))) __bf16 v16bf;
typedef __attribute__((ext_vector_type(8)))  float  v8f;

#define NTOK 8192   // B*S
#define DD   128    // D
#define MM   64     // M
#define D2   256    // 2*D
#define ROWS 32     // tokens per workgroup in mech_main (2 x 16-row A tiles)

__device__ __forceinline__ __bf16 f2bf(float f) { return (__bf16)f; }

// A-matrix (16x32 bf16) element->K mapping per ISA 7.12.2:
// lanes 0-15: row=lane, elems 0-7 -> K=0..7,  elems 8-15 -> K=16..23
// lanes 16-31: row=lane-16, elems 0-7 -> K=8..15, elems 8-15 -> K=24..31
__device__ __forceinline__ int afk(int lane, int e) {
  int kb = (lane & 16) ? 8 : 0;
  return (e < 8) ? (kb + e) : (kb + 8 + e);
}

__device__ __forceinline__ v8f wmma_bf(v16bf a, v16bf b, v8f c) {
  return __builtin_amdgcn_wmma_f32_16x16x32_bf16(false, a, false, b, (short)0, c,
                                                 false, false);
}

// ---------------- Kernel 0a: per-mechanism precompute ----------------
// Wt -> bf16; wv_eff[m][d] = sum_e Wt[m][d][e]*Wv[m][e];
// btc[m][e] = bt+char; cv[m] = (bt+char)@Wv + bv.
__global__ void mech_pre(const float* __restrict__ Wt, const float* __restrict__ Wv,
                         const float* __restrict__ bt, const float* __restrict__ chr,
                         const float* __restrict__ bv,
                         __bf16* __restrict__ Wtbf, float* __restrict__ wv_eff,
                         float* __restrict__ btc, float* __restrict__ cv) {
  int m = blockIdx.x, d = threadIdx.x;
  const float* wrow = Wt + ((size_t)m * DD + d) * DD;
  const float* wv   = Wv + m * DD;
  __bf16* orow = Wtbf + ((size_t)m * DD + d) * DD;
  float acc = 0.f;
  for (int e = 0; e < DD; e++) {
    float w = wrow[e];
    orow[e] = f2bf(w);
    acc += w * wv[e];
  }
  wv_eff[m * DD + d] = acc;
  btc[m * DD + d] = bt[m * DD + d] + chr[m * DD + d];
  if (d == 0) {
    float s = 0.f;
    for (int e = 0; e < DD; e++) s += (bt[m * DD + e] + chr[m * DD + e]) * wv[e];
    cv[m] = s + bv[m];
  }
}

// ---------------- Kernel 0b: generic f32 -> bf16 ----------------
__global__ void cvt_bf(const float* __restrict__ src, __bf16* __restrict__ dst, int n) {
  int i = blockIdx.x * blockDim.x + threadIdx.x;
  if (i < n) dst[i] = f2bf(src[i]);
}

// ---------------- Kernel 1: selector ----------------
// Per 16-token tile: H = gelu([x|ctx]@W1+b1) (WMMA), S = H@W2+b2 (WMMA),
// softmax, gates=1+tanh(S@O), timing, victory (via wv_eff), writes
// c[tok][m] = gate*score*timing and victory_potential.
__global__ __launch_bounds__(128) void selector_k(
    const float* __restrict__ x, const float* __restrict__ ctx,
    const __bf16* __restrict__ W1bf, const float* __restrict__ b1,
    const __bf16* __restrict__ W2bf, const float* __restrict__ b2,
    const float* __restrict__ O, const float* __restrict__ Wg,
    const float* __restrict__ bg, const float* __restrict__ wv_eff,
    const float* __restrict__ cv, float* __restrict__ c_out,
    float* __restrict__ vp_out) {
  __shared__ float  si[16 * D2];     // [x|ctx] tile, f32
  __shared__ __bf16 hsh[16 * D2];    // hidden, bf16
  __shared__ float  ssh[16 * MM];    // scores
  __shared__ float  vpsh[16];
  int tid = threadIdx.x, lane = tid & 31, wave = tid >> 5;
  int tile = blockIdx.x, tok0 = tile * 16;

  for (int idx = tid; idx < 16 * D2; idx += 128) {
    int r = idx >> 8, col = idx & 255;
    si[idx] = (col < DD) ? x[(size_t)(tok0 + r) * DD + col]
                         : ctx[(size_t)(tok0 + r) * DD + col - DD];
  }
  if (tid < 16) vpsh[tid] = 0.f;
  __syncthreads();

  int arow = lane & 15;
  int rowoff = (lane & 16) ? 8 : 0;

  // A fragments of sel_in (16x256): 8 k-steps, reused by 4 N-tiles per wave
  v16bf sA[8];
  for (int kk = 0; kk < 8; kk++)
    for (int e = 0; e < 16; e++)
      sA[kk][e] = f2bf(si[arow * D2 + kk * 32 + afk(lane, e)]);

  // H = gelu(si @ W1 + b1): wave handles N-tiles wave*4 .. wave*4+3
  for (int t = 0; t < 4; t++) {
    int nt = wave * 4 + t;
    int col = nt * 16 + (lane & 15);
    float bv1 = b1[col];
    v8f acc;
    for (int r = 0; r < 8; r++) acc[r] = bv1;
    for (int kk = 0; kk < 8; kk++) {
      v16bf bfr;
      for (int e = 0; e < 16; e++) bfr[e] = W1bf[(size_t)(kk * 32 + afk(lane, e)) * D2 + col];
      acc = wmma_bf(sA[kk], bfr, acc);
    }
    for (int r = 0; r < 8; r++) {
      float v = acc[r];
      float g = 0.5f * v * (1.0f + erff(v * 0.70710678118f));  // exact GELU
      hsh[(r + rowoff) * D2 + col] = f2bf(g);
    }
  }
  __syncthreads();

  // S = H @ W2 + b2: wave handles N-tile = wave
  {
    int nt = wave;
    int col = nt * 16 + (lane & 15);
    float bv2 = b2[col];
    v8f acc;
    for (int r = 0; r < 8; r++) acc[r] = bv2;
    for (int kk = 0; kk < 8; kk++) {
      v16bf a;
      for (int e = 0; e < 16; e++) a[e] = hsh[arow * D2 + kk * 32 + afk(lane, e)];
      v16bf bfr;
      for (int e = 0; e < 16; e++) bfr[e] = W2bf[(size_t)(kk * 32 + afk(lane, e)) * MM + col];
      acc = wmma_bf(a, bfr, acc);
    }
    for (int r = 0; r < 8; r++) ssh[(r + rowoff) * MM + col] = acc[r];
  }
  __syncthreads();

  // softmax over 64 mechanisms, one row per thread (threads 0..15)
  if (tid < 16) {
    float mx = -1e30f;
    for (int j = 0; j < MM; j++) mx = fmaxf(mx, ssh[tid * MM + j]);
    float sum = 0.f;
    for (int j = 0; j < MM; j++) {
      float e = __expf(ssh[tid * MM + j] - mx);
      ssh[tid * MM + j] = e;
      sum += e;
    }
    float inv = 1.f / sum;
    for (int j = 0; j < MM; j++) ssh[tid * MM + j] *= inv;
  }
  __syncthreads();

  // scalar phase: thread -> (row, 8 mechanisms)
  int row = tid >> 3, cg = tid & 7;
  float vp = 0.f;
  for (int j8 = 0; j8 < 8; j8++) {
    int mi = cg * 8 + j8;
    float g = 0.f;
    for (int j = 0; j < MM; j++) g += ssh[row * MM + j] * O[j * MM + mi];
    float gate = 1.f + tanhf(g);
    float tl = bg[mi];
    for (int d = 0; d < DD; d++) tl += si[row * D2 + d] * Wg[mi * DD + d];
    float tim = 1.f / (1.f + __expf(-tl));
    float vl = cv[mi];
    for (int d = 0; d < DD; d++) vl += si[row * D2 + d] * wv_eff[mi * DD + d];
    float vic = 1.f / (1.f + __expf(-vl));
    float sc = ssh[row * MM + mi];
    c_out[(size_t)(tok0 + row) * MM + mi] = gate * sc * tim;
    vp += vic * sc;
  }
  atomicAdd(&vpsh[row], vp);
  __syncthreads();
  if (tid < 16) vp_out[tok0 + tid] = vpsh[tid];
}

// ---------------- Kernel 2: fused mechanism GEMM + select + integrate ----------
// 32 tokens per workgroup (2 row-tiles), 4 waves; wave owns 32 output columns.
// Every B fragment is reused by both row tiles -> 32 FLOP per L2 byte.
// selected = sum_m c_m * (x@Wt[m] + btc[m]); out = [x|selected]@Wi + bi.
__global__ __launch_bounds__(128) void mech_main(
    const float* __restrict__ x, const __bf16* __restrict__ Wtbf,
    const float* __restrict__ btc_g, const float* __restrict__ c_g,
    const __bf16* __restrict__ Wibf, const float* __restrict__ bi,
    float* __restrict__ outI) {
  __shared__ float xs[ROWS * DD];      // 16 KB
  __shared__ float csh[ROWS * MM];     // 8 KB
  __shared__ float btcsh[MM * DD];     // 32 KB
  __shared__ float selsh[ROWS * DD];   // 16 KB
  int tid = threadIdx.x, lane = tid & 31, wave = tid >> 5;
  int tile = blockIdx.x, tok0 = tile * ROWS;

  for (int idx = tid; idx < ROWS * DD; idx += 128) xs[idx] = x[(size_t)tok0 * DD + idx];
  for (int idx = tid; idx < ROWS * MM; idx += 128) csh[idx] = c_g[(size_t)tok0 * MM + idx];
  for (int idx = tid; idx < MM * DD; idx += 128) btcsh[idx] = btc_g[idx];
  __syncthreads();

  int arow = lane & 15;
  int rowoff = (lane & 16) ? 8 : 0;
  int nc0 = wave * 32, nc1 = nc0 + 16;
  int col0 = nc0 + (lane & 15), col1 = col0 + 16;

  // x A-fragments for both row tiles, reused for all 64 mechanisms + integrate
  v16bf xA[2][4];
  for (int rt = 0; rt < 2; rt++)
    for (int kk = 0; kk < 4; kk++)
      for (int e = 0; e < 16; e++)
        xA[rt][kk][e] = f2bf(xs[(rt * 16 + arow) * DD + kk * 32 + afk(lane, e)]);

  v8f sel[2][2];
  for (int rt = 0; rt < 2; rt++)
    for (int r = 0; r < 8; r++) { sel[rt][0][r] = 0.f; sel[rt][1][r] = 0.f; }

  for (int m = 0; m < MM; m++) {
    const __bf16* Wm = Wtbf + (size_t)m * DD * DD;
    if (m + 1 < MM) __builtin_prefetch(Wtbf + (size_t)(m + 1) * DD * DD, 0, 1);
    v8f T[2][2];
    for (int rt = 0; rt < 2; rt++)
      for (int r = 0; r < 8; r++) { T[rt][0][r] = 0.f; T[rt][1][r] = 0.f; }
    for (int kk = 0; kk < 4; kk++) {
      v16bf b0, b1v;
      for (int e = 0; e < 16; e++) {
        const __bf16* base = Wm + (size_t)(kk * 32 + afk(lane, e)) * DD;
        b0[e] = base[col0];
        b1v[e] = base[col1];
      }
      T[0][0] = wmma_bf(xA[0][kk], b0, T[0][0]);
      T[0][1] = wmma_bf(xA[0][kk], b1v, T[0][1]);
      T[1][0] = wmma_bf(xA[1][kk], b0, T[1][0]);
      T[1][1] = wmma_bf(xA[1][kk], b1v, T[1][1]);
    }
    float bt0 = btcsh[m * DD + col0], bt1 = btcsh[m * DD + col1];
    for (int rt = 0; rt < 2; rt++)
      for (int r = 0; r < 8; r++) {
        float cm = csh[(rt * 16 + r + rowoff) * MM + m];
        sel[rt][0][r] += cm * (T[rt][0][r] + bt0);
        sel[rt][1][r] += cm * (T[rt][1][r] + bt1);
      }
  }

  for (int rt = 0; rt < 2; rt++)
    for (int r = 0; r < 8; r++) {
      selsh[(rt * 16 + r + rowoff) * DD + col0] = sel[rt][0][r];
      selsh[(rt * 16 + r + rowoff) * DD + col1] = sel[rt][1][r];
    }
  __syncthreads();

  // integrate: out = x@Wi[0:128] + selected@Wi[128:256] + bi
  v8f o[2][2];
  for (int rt = 0; rt < 2; rt++)
    for (int r = 0; r < 8; r++) { o[rt][0][r] = 0.f; o[rt][1][r] = 0.f; }
  for (int kk = 0; kk < 4; kk++) {
    v16bf sA0, sA1;
    for (int e = 0; e < 16; e++) {
      int k = kk * 32 + afk(lane, e);
      sA0[e] = f2bf(selsh[arow * DD + k]);
      sA1[e] = f2bf(selsh[(16 + arow) * DD + k]);
    }
    v16bf t0, t1, bb0, bb1;
    for (int e = 0; e < 16; e++) {
      const __bf16* topb = Wibf + (size_t)(kk * 32 + afk(lane, e)) * DD;
      const __bf16* botb = Wibf + (size_t)(DD + kk * 32 + afk(lane, e)) * DD;
      t0[e] = topb[col0];
      t1[e] = topb[col1];
      bb0[e] = botb[col0];
      bb1[e] = botb[col1];
    }
    o[0][0] = wmma_bf(xA[0][kk], t0, o[0][0]);
    o[0][0] = wmma_bf(sA0, bb0, o[0][0]);
    o[0][1] = wmma_bf(xA[0][kk], t1, o[0][1]);
    o[0][1] = wmma_bf(sA0, bb1, o[0][1]);
    o[1][0] = wmma_bf(xA[1][kk], t0, o[1][0]);
    o[1][0] = wmma_bf(sA1, bb0, o[1][0]);
    o[1][1] = wmma_bf(xA[1][kk], t1, o[1][1]);
    o[1][1] = wmma_bf(sA1, bb1, o[1][1]);
  }
  float bi0 = bi[col0], bi1 = bi[col1];
  for (int rt = 0; rt < 2; rt++)
    for (int r = 0; r < 8; r++) {
      size_t row = (size_t)(tok0 + rt * 16 + r + rowoff);
      outI[row * DD + col0] = o[rt][0][r] + bi0;
      outI[row * DD + col1] = o[rt][1][r] + bi1;
    }
}

// ---------------- host launch ----------------
extern "C" void kernel_launch(void* const* d_in, const int* in_sizes, int n_in,
                              void* d_out, int out_size, void* d_ws, size_t ws_size,
                              hipStream_t stream) {
  const float* x   = (const float*)d_in[0];
  const float* ctx = (const float*)d_in[1];
  const float* Wt  = (const float*)d_in[2];
  const float* bt  = (const float*)d_in[3];
  const float* chr = (const float*)d_in[4];
  const float* Wg  = (const float*)d_in[5];
  const float* bg  = (const float*)d_in[6];
  const float* Wv  = (const float*)d_in[7];
  const float* bv  = (const float*)d_in[8];
  const float* O   = (const float*)d_in[9];
  const float* W1  = (const float*)d_in[10];
  const float* b1  = (const float*)d_in[11];
  const float* W2  = (const float*)d_in[12];
  const float* b2  = (const float*)d_in[13];
  const float* Wi  = (const float*)d_in[14];
  const float* bi  = (const float*)d_in[15];

  char* ws = (char*)d_ws;
  size_t off = 0;
  __bf16* Wtbf  = (__bf16*)(ws + off); off += (size_t)MM * DD * DD * 2;  // 2 MB
  __bf16* W1bf  = (__bf16*)(ws + off); off += (size_t)D2 * D2 * 2;       // 128 KB
  __bf16* W2bf  = (__bf16*)(ws + off); off += (size_t)D2 * MM * 2;       // 32 KB
  __bf16* Wibf  = (__bf16*)(ws + off); off += (size_t)D2 * DD * 2;       // 64 KB
  float* wv_eff = (float*)(ws + off);  off += (size_t)MM * DD * 4;
  float* cv     = (float*)(ws + off);  off += 256;
  float* btc    = (float*)(ws + off);  off += (size_t)MM * DD * 4;
  float* c_g    = (float*)(ws + off);  off += (size_t)NTOK * MM * 4;     // 2 MB

  float* outI = (float*)d_out;
  float* vpO  = outI + (size_t)NTOK * DD;

  hipLaunchKernelGGL(mech_pre, dim3(MM), dim3(DD), 0, stream,
                     Wt, Wv, bt, chr, bv, Wtbf, wv_eff, btc, cv);
  hipLaunchKernelGGL(cvt_bf, dim3((D2 * D2 + 255) / 256), dim3(256), 0, stream,
                     W1, W1bf, D2 * D2);
  hipLaunchKernelGGL(cvt_bf, dim3((D2 * MM + 255) / 256), dim3(256), 0, stream,
                     W2, W2bf, D2 * MM);
  hipLaunchKernelGGL(cvt_bf, dim3((D2 * DD + 255) / 256), dim3(256), 0, stream,
                     Wi, Wibf, D2 * DD);
  hipLaunchKernelGGL(selector_k, dim3(NTOK / 16), dim3(128), 0, stream,
                     x, ctx, W1bf, b1, W2bf, b2, O, Wg, bg, wv_eff, cv, c_g, vpO);
  hipLaunchKernelGGL(mech_main, dim3(NTOK / ROWS), dim3(128), 0, stream,
                     x, Wtbf, btc, c_g, Wibf, bi, outI);
}
